// PerformerBasicMHA_86260123173284
// MI455X (gfx1250) — matrix operand order repaired and checked by reading, not compiled, not run on hardware
//
#include <hip/hip_runtime.h>

// ---------------------------------------------------------------------------
// Performer linear attention for MI455X (gfx1250, wave32, WMMA f16 -> f32,
// TDM tensor_load_to_lds for activation tiles in the big QKV GEMM)
// ---------------------------------------------------------------------------
#define EPSV 1e-3f

typedef __attribute__((ext_vector_type(16))) _Float16 v16h;
typedef __attribute__((ext_vector_type(8)))  float    v8f;
typedef __attribute__((ext_vector_type(4)))  unsigned int u32x4;
typedef __attribute__((ext_vector_type(4)))  int      i32x4;
typedef __attribute__((ext_vector_type(8)))  int      i32x8;

union HFrag { v16h v; uint4 u[2]; };          // one WMMA A/B fragment (16 halves)
union H8   { _Float16 h[8];  uint4 u;    };
union H16  { _Float16 h[16]; uint4 u[2]; };

constexpr int NB = 4, S = 4096, E = 1024, H = 16, D = 64;
constexpr int MTOT = NB * S;                  // 16384 rows of x
constexpr int BM = 128, BN = 128, BK = 32;    // stage-1 GEMM tile
constexpr int LDA = BK + 8;                   // LDS row stride (halves), padded

__device__ __forceinline__ v8f wmma_f16(const HFrag& a, const HFrag& b, v8f c) {
  return __builtin_amdgcn_wmma_f32_16x16x32_f16(false, a.v, false, b.v,
                                                (short)0, c, false, false);
}

// generic -> LDS(AS3) byte offset for TDM descriptors
typedef __attribute__((address_space(3))) const char* lds_cptr_t;
__device__ __forceinline__ unsigned lds_base(const void* p) {
  return (unsigned)(unsigned long long)(lds_cptr_t)p;
}

// ---------------------------------------------------------------------------
// Stage 0: x (f32) -> f16
// ---------------------------------------------------------------------------
__global__ __launch_bounds__(256) void cvt_f32_to_f16(const float* __restrict__ in,
                                                      _Float16* __restrict__ out,
                                                      int n) {
  int i = (blockIdx.x * 256 + threadIdx.x) * 8;
  if (i + 8 <= n) {
    float4 a = *(const float4*)(in + i);
    float4 b = *(const float4*)(in + i + 4);
    H8 p;
    p.h[0] = (_Float16)a.x; p.h[1] = (_Float16)a.y;
    p.h[2] = (_Float16)a.z; p.h[3] = (_Float16)a.w;
    p.h[4] = (_Float16)b.x; p.h[5] = (_Float16)b.y;
    p.h[6] = (_Float16)b.z; p.h[7] = (_Float16)b.w;
    *(uint4*)(out + i) = p.u;
  }
}

// ---------------------------------------------------------------------------
// Stage 1: C = x @ W^T + b  (M=16384, N=1024, K=1024), f16 WMMA, f32 acc.
//  A (x) tiles are fetched by the Tensor Data Mover straight into padded LDS
//  (tile 32x128 halves; pad_interval=3 -> every 16 DWORDs (=one 64B row),
//   pad_amount=3 -> 4 DWORDs, giving exactly the LDA=40-half row stride).
//  B (W) tiles are loaded manually because they need f32->f16 conversion.
//  RELU : apply relu()+EPS (Q and K)
//  TRANS: store transposed per head as (n,h,d,s) f16 (K and V); else (n,s,E).
// ---------------------------------------------------------------------------
template <int RELU, int TRANS>
__global__ __launch_bounds__(256) void qkv_gemm(const _Float16* __restrict__ xh,
                                                const float* __restrict__ W,
                                                const float* __restrict__ bias,
                                                _Float16* __restrict__ outp) {
  __shared__ __align__(16) _Float16 At[2][BM * LDA];
  __shared__ __align__(16) _Float16 Bt[2][BN * LDA];

  const int tid  = threadIdx.x;
  const int lane = tid & 31;
  const int w    = tid >> 5;
  const int wm   = w & 1;      // 2 M-subtiles of 64
  const int wn   = w >> 1;     // 4 N-subtiles of 32
  const int m0   = blockIdx.x * BM;
  const int n0   = blockIdx.y * BN;

  const int rowL = tid >> 1;          // 128 rows, 2 threads/row
  const int colL = (tid & 1) * 16;    // 16 f32 each (B tile)

  // ---- TDM: async A-tile (128 x 32 halves) -> LDS, issued by wave 0 only ----
  auto issue_tdm_A = [&](int k0, int buf) {
    if (tid < 32) {
      const unsigned long long ga =
          (unsigned long long)xh + (((size_t)m0 * E + k0) * 2);
      u32x4 g0;
      g0.x = 1u;                                        // count=1, user mode
      g0.y = lds_base(&At[buf][0]);                     // lds_addr (bytes)
      g0.z = (unsigned)(ga & 0xFFFFFFFFull);            // global_addr[31:0]
      g0.w = (unsigned)((ga >> 32) & 0x01FFFFFFull)     // global_addr[56:32]
             | (2u << 30);                              // type=2 ("image")
      // word0: data_size=1(2B) | pad_enable | pad_interval=3 | pad_amount=3
      const i32x8 g1 = {
          (int)((1u << 16) | (1u << 20) | (3u << 22) | (3u << 25)),
          (int)((unsigned)E << 16),                 // tensor_dim0[15:0]
          (int)(((unsigned)MTOT & 0xFFFFu) << 16),  // tensor_dim1[15:0]
          (int)((unsigned)BK << 16),                // tile_dim0 = 32
          (int)BM,                                  // tile_dim1 = 128
          (int)E,                                   // tensor_dim0_stride
          0, 0};
      const i32x4 z4 = {0, 0, 0, 0};
#if __clang_major__ >= 23
      const i32x8 z8 = {0, 0, 0, 0, 0, 0, 0, 0};
      __builtin_amdgcn_tensor_load_to_lds(g0, g1, z4, z4, z8, 0);
#else
      __builtin_amdgcn_tensor_load_to_lds(g0, g1, z4, z4, 0);
#endif
    }
  };

  // ---- manual B-tile load (f32 -> f16) ----
  auto load_B = [&](int k0, int buf) {
    const float* bg = W + (size_t)(n0 + rowL) * E + k0 + colL;
    float4 b0 = *(const float4*)(bg + 0);
    float4 b1 = *(const float4*)(bg + 4);
    float4 b2 = *(const float4*)(bg + 8);
    float4 b3 = *(const float4*)(bg + 12);
    H16 p;
    p.h[0]  = (_Float16)b0.x; p.h[1]  = (_Float16)b0.y;
    p.h[2]  = (_Float16)b0.z; p.h[3]  = (_Float16)b0.w;
    p.h[4]  = (_Float16)b1.x; p.h[5]  = (_Float16)b1.y;
    p.h[6]  = (_Float16)b1.z; p.h[7]  = (_Float16)b1.w;
    p.h[8]  = (_Float16)b2.x; p.h[9]  = (_Float16)b2.y;
    p.h[10] = (_Float16)b2.z; p.h[11] = (_Float16)b2.w;
    p.h[12] = (_Float16)b3.x; p.h[13] = (_Float16)b3.y;
    p.h[14] = (_Float16)b3.z; p.h[15] = (_Float16)b3.w;
    *(uint4*)&Bt[buf][rowL * LDA + colL]     = p.u[0];
    *(uint4*)&Bt[buf][rowL * LDA + colL + 8] = p.u[1];
    __builtin_prefetch(bg + BK, 0, 0);   // global_prefetch_b8 next k-tile
  };

  const v8f vzero = {0.f, 0.f, 0.f, 0.f, 0.f, 0.f, 0.f, 0.f};
  v8f acc[4][2];
#pragma unroll
  for (int i = 0; i < 4; ++i)
#pragma unroll
    for (int j = 0; j < 2; ++j) acc[i][j] = vzero;

  constexpr int NK = E / BK;
  issue_tdm_A(0, 0);
  load_B(0, 0);
  if (tid < 32) __builtin_amdgcn_s_wait_tensorcnt(0);
  __syncthreads();

  const int ar    = lane & 15;
  const int khalf = (lane >> 4) * 8;
  const int kb16  = (lane >> 4) * 16;

  for (int kk = 0; kk < NK; ++kk) {
    const int buf = kk & 1;
    if (kk + 1 < NK) {
      issue_tdm_A((kk + 1) * BK, buf ^ 1);
      load_B((kk + 1) * BK, buf ^ 1);
    }

    HFrag af[4], bf[2];
#pragma unroll
    for (int mt = 0; mt < 4; ++mt) {
      const _Float16* p = &At[buf][(wm * 64 + mt * 16 + ar) * LDA + kb16];
      af[mt].u[0] = *(const uint4*)p;
      af[mt].u[1] = *(const uint4*)(p + 8);
    }
#pragma unroll
    for (int nt = 0; nt < 2; ++nt) {
      const _Float16* p = &Bt[buf][(wn * 32 + nt * 16 + ar) * LDA + kb16];
      bf[nt].u[0] = *(const uint4*)p;
      bf[nt].u[1] = *(const uint4*)(p + 8);
    }
#pragma unroll
    for (int mt = 0; mt < 4; ++mt)
#pragma unroll
      for (int nt = 0; nt < 2; ++nt)
        acc[mt][nt] = wmma_f16(af[mt], bf[nt], acc[mt][nt]);

    if (tid < 32 && kk + 1 < NK) __builtin_amdgcn_s_wait_tensorcnt(0);
    __syncthreads();
  }

  // Epilogue: bias (+relu+eps); D layout: lane = column n, 8 consecutive rows m.
#pragma unroll
  for (int nt = 0; nt < 2; ++nt) {
    const int ncol = n0 + wn * 32 + nt * 16 + ar;
    const float bb = bias[ncol];
#pragma unroll
    for (int mt = 0; mt < 4; ++mt) {
      const int mbase = m0 + wm * 64 + mt * 16 + (lane >> 4) * 8;
      if (TRANS) {
        // store K/V transposed: (n, h, d, s) — 8 consecutive tokens -> b128
        H8 pk;
#pragma unroll
        for (int v = 0; v < 8; ++v) {
          float f = acc[mt][nt][v] + bb;
          if (RELU) f = fmaxf(f, 0.f) + EPSV;
          pk.h[v] = (_Float16)f;
        }
        const int hidx = ncol >> 6, didx = ncol & 63;
        const int nidx = mbase >> 12, sidx = mbase & (S - 1);
        _Float16* dst = outp + (((size_t)(nidx * H + hidx) * D + didx) * S + sidx);
        *(uint4*)dst = pk.u;
      } else {
        // Q row-major (n, s, E)
#pragma unroll
        for (int v = 0; v < 8; ++v) {
          float f = acc[mt][nt][v] + bb;
          if (RELU) f = fmaxf(f, 0.f) + EPSV;
          outp[(size_t)(mbase + v) * E + ncol] = (_Float16)f;
        }
      }
    }
  }
}

// ---------------------------------------------------------------------------
// Stage 2: per (n,h): kvT[e][d] = sum_s V[s,e]*K[s,d]  (A=Vtr, B=Ktr, K-dim=s)
// plus ksum[d] = sum_s K[s,d]. 8 waves: 2 (M=e) x 4 s-chunks of 1024.
// Cross-wave reduction through LDS f32 atomics.
// ---------------------------------------------------------------------------
__global__ __launch_bounds__(256) void kv_ksum_kernel(const _Float16* __restrict__ Ktr,
                                                      const _Float16* __restrict__ Vtr,
                                                      _Float16* __restrict__ kvT,
                                                      float* __restrict__ ksum) {
  __shared__ float kvbuf[D * D];
  __shared__ float ksbuf[D];
  const int tid = threadIdx.x;
  const int nh  = blockIdx.x;
  const _Float16* Kb = Ktr + (size_t)nh * D * S;
  const _Float16* Vb = Vtr + (size_t)nh * D * S;

  for (int i = tid; i < D * D; i += 256) kvbuf[i] = 0.f;
  if (tid < D) ksbuf[tid] = 0.f;
  __syncthreads();

  const int lane = tid & 31;
  const int w    = tid >> 5;
  const int wm   = w & 1;
  const int s0   = (w >> 1) * (S / 4);
  const int ar    = lane & 15;
  const int khalf = (lane >> 4) * 8;
  const int kb16  = (lane >> 4) * 16;

  const v8f vzero = {0.f, 0.f, 0.f, 0.f, 0.f, 0.f, 0.f, 0.f};
  v8f acc[2][4];
#pragma unroll
  for (int i = 0; i < 2; ++i)
#pragma unroll
    for (int j = 0; j < 4; ++j) acc[i][j] = vzero;

  for (int si = 0; si < S / 4; si += 32) {
    const int s = s0 + si;
    HFrag af[2], bf[4];
#pragma unroll
    for (int mt = 0; mt < 2; ++mt) {     // A = Vtr rows (e), contiguous along s
      const _Float16* p = Vb + (size_t)(wm * 32 + mt * 16 + ar) * S + s + kb16;
      af[mt].u[0] = *(const uint4*)p;
      af[mt].u[1] = *(const uint4*)(p + 8);
    }
#pragma unroll
    for (int nt = 0; nt < 4; ++nt) {     // B = Ktr rows (d), contiguous along s
      const _Float16* p = Kb + (size_t)(nt * 16 + ar) * S + s + kb16;
      bf[nt].u[0] = *(const uint4*)p;
      bf[nt].u[1] = *(const uint4*)(p + 8);
    }
#pragma unroll
    for (int mt = 0; mt < 2; ++mt)
#pragma unroll
      for (int nt = 0; nt < 4; ++nt)
        acc[mt][nt] = wmma_f16(af[mt], bf[nt], acc[mt][nt]);
  }

#pragma unroll
  for (int mt = 0; mt < 2; ++mt)
#pragma unroll
    for (int nt = 0; nt < 4; ++nt)
#pragma unroll
      for (int v = 0; v < 8; ++v) {
        const int m = wm * 32 + mt * 16 + (lane >> 4) * 8 + v;  // e
        const int n = nt * 16 + ar;                             // d
        atomicAdd(&kvbuf[m * D + n], acc[mt][nt][v]);
      }

  { // ksum: row sums of Ktr
    const int d = tid & 63, q = tid >> 6;
    const _Float16* p = Kb + (size_t)d * S + q * (S / 4);
    float sum = 0.f;
    for (int j = 0; j < S / 4; j += 8) {
      H8 t; t.u = *(const uint4*)(p + j);
#pragma unroll
      for (int v = 0; v < 8; ++v) sum += (float)t.h[v];
    }
    atomicAdd(&ksbuf[d], sum);
  }
  __syncthreads();

  _Float16* kvo = kvT + (size_t)nh * D * D;     // kvT[e][d] row-major
  for (int i = tid; i < D * D; i += 256) kvo[i] = (_Float16)kvbuf[i];
  if (tid < D) ksum[nh * D + tid] = ksbuf[tid];
}

// ---------------------------------------------------------------------------
// Stage 3: per (n,h, 128-token tile): D[e][s] = sum_d kvT[e][d] * Q[s][d],
// scaled by 1/(q . ksum) per token, transposed-stored as out[(n,h,s,d)] (f32).
// ---------------------------------------------------------------------------
constexpr int ST = 128;
constexpr int LQ = D + 8;   // padded LDS stride (halves)

__global__ __launch_bounds__(256) void attn_out_kernel(const _Float16* __restrict__ Qh,
                                                       const _Float16* __restrict__ kvT,
                                                       const float* __restrict__ ksum,
                                                       float* __restrict__ out) {
  __shared__ __align__(16) _Float16 kvs[D * LQ];
  __shared__ __align__(16) _Float16 qs[ST * LQ];
  __shared__ float kss[D];
  __shared__ float dinv[ST];

  const int tid  = threadIdx.x;
  const int nh   = blockIdx.y;
  const int nidx = nh >> 4;
  const int hh   = nh & 15;
  const int s0   = blockIdx.x * ST;

  { // kvT tile: 64x64 halves
    const uint4* src = (const uint4*)(kvT + (size_t)nh * D * D);
    for (int j = tid; j < D * D / 8; j += 256) {
      const int row = j >> 3, c8 = (j & 7) * 8;
      *(uint4*)&kvs[row * LQ + c8] = src[j];
    }
  }
  if (tid < D) kss[tid] = ksum[nh * D + tid];
  { // Q tile: 128 tokens x 64 halves (head slice)
    for (int j = tid; j < ST * D / 8; j += 256) {
      const int row = j >> 3, c8 = (j & 7) * 8;
      const _Float16* src =
          Qh + (size_t)(nidx * S + s0 + row) * E + hh * D + c8;
      *(uint4*)&qs[row * LQ + c8] = *(const uint4*)src;
    }
  }
  __syncthreads();

  if (tid < ST) {  // diag = q . ksum  (q >= eps, ksum > 0 => no div-by-zero)
    float sum = 0.f;
#pragma unroll 8
    for (int d = 0; d < D; ++d) sum += (float)qs[tid * LQ + d] * kss[d];
    dinv[tid] = 1.0f / sum;
  }
  __syncthreads();

  const int lane  = tid & 31;
  const int w     = tid >> 5;          // wave -> 16-token group
  const int ar    = lane & 15;
  const int khalf = (lane >> 4) * 8;
  const int kb16  = (lane >> 4) * 16;

  const v8f vzero = {0.f, 0.f, 0.f, 0.f, 0.f, 0.f, 0.f, 0.f};
  v8f acc[4];
#pragma unroll
  for (int i = 0; i < 4; ++i) acc[i] = vzero;

#pragma unroll
  for (int kstep = 0; kstep < 2; ++kstep) {
    HFrag bfr;   // B[k=d][n=token] = Q[token][d] : contiguous per lane
    const _Float16* pb = &qs[(w * 16 + ar) * LQ + kstep * 32 + kb16];
    bfr.u[0] = *(const uint4*)pb;
    bfr.u[1] = *(const uint4*)(pb + 8);
#pragma unroll
    for (int mt = 0; mt < 4; ++mt) {
      HFrag afr; // A[m=e][k=d] = kvT row, contiguous
      const _Float16* pa = &kvs[(mt * 16 + ar) * LQ + kstep * 32 + kb16];
      afr.u[0] = *(const uint4*)pa;
      afr.u[1] = *(const uint4*)(pa + 8);
      acc[mt] = wmma_f16(afr, bfr, acc[mt]);
    }
  }

  // per lane: fixed token, 8 consecutive e per acc -> contiguous f32 stores
  const int tl = w * 16 + ar;
  const float di = dinv[tl];
  float* obase = out + ((size_t)nh * S + (s0 + tl)) * D;
#pragma unroll
  for (int mt = 0; mt < 4; ++mt) {
    const int e0 = mt * 16 + (lane >> 4) * 8;
    float4 f0 = make_float4(acc[mt][0] * di, acc[mt][1] * di,
                            acc[mt][2] * di, acc[mt][3] * di);
    float4 f1 = make_float4(acc[mt][4] * di, acc[mt][5] * di,
                            acc[mt][6] * di, acc[mt][7] * di);
    *(float4*)(obase + e0)     = f0;
    *(float4*)(obase + e0 + 4) = f1;
  }
}

// ---------------------------------------------------------------------------
extern "C" void kernel_launch(void* const* d_in, const int* in_sizes, int n_in,
                              void* d_out, int out_size, void* d_ws, size_t ws_size,
                              hipStream_t stream) {
  (void)in_sizes; (void)n_in; (void)out_size; (void)ws_size;
  const float* x  = (const float*)d_in[0];
  const float* wq = (const float*)d_in[1];
  const float* bq = (const float*)d_in[2];
  const float* wk = (const float*)d_in[3];
  const float* bk = (const float*)d_in[4];
  const float* wv = (const float*)d_in[5];
  const float* bv = (const float*)d_in[6];

  char* ws = (char*)d_ws;
  _Float16* xh   = (_Float16*)(ws);                               // 32 MB
  _Float16* Qh   = (_Float16*)(ws + (size_t)33554432);            // 32 MB
  _Float16* Ktr  = (_Float16*)(ws + (size_t)67108864);            // 32 MB
  _Float16* Vtr  = (_Float16*)(ws + (size_t)100663296);           // 32 MB
  _Float16* kvTp = (_Float16*)(ws + (size_t)134217728);           // 512 KB
  float*    ksum = (float*)   (ws + (size_t)134217728 + 524288);  // 16 KB

  const int total = NB * S * E;  // 16,777,216
  cvt_f32_to_f16<<<total / (256 * 8), 256, 0, stream>>>(x, xh, total);

  dim3 g1(MTOT / BM, E / BN);    // (128, 8)
  qkv_gemm<1, 0><<<g1, 256, 0, stream>>>(xh, wq, bq, Qh);   // Q: relu+eps, row-major
  qkv_gemm<1, 1><<<g1, 256, 0, stream>>>(xh, wk, bk, Ktr);  // K: relu+eps, (n,h,d,s)
  qkv_gemm<0, 1><<<g1, 256, 0, stream>>>(xh, wv, bv, Vtr);  // V: plain,    (n,h,d,s)

  kv_ksum_kernel<<<NB * H, 256, 0, stream>>>(Ktr, Vtr, kvTp, ksum);

  attn_out_kernel<<<dim3(S / ST, NB * H), 256, 0, stream>>>(Qh, kvTp, ksum,
                                                            (float*)d_out);
}
